// MoE_out_43069932044306
// MI455X (gfx1250) — matrix operand order repaired and checked by reading, not compiled, not run on hardware
//
#include <hip/hip_runtime.h>

// ---------------------------------------------------------------------------
// MoE dense-all-experts, fused bf16-WMMA implementation for gfx1250 (CDNA5).
// B=4, S=2048, D=1024, E=8, H=2048  ->  T = 8192 tokens.
// Round 3: N-dimension register blocking widened to 4 col-tiles per K-pass:
//   0.5 ds_load_b128 + 1 global_load_b128 per WMMA (LDS ~128 B/clk/WGP).
// ---------------------------------------------------------------------------

#define Dd   1024
#define Hh   2048
#define Ee   8
#define Tt   8192
#define TM   32          // tokens per block
#define NTHR 512         // 16 waves

typedef __bf16 bf16_t;
typedef __attribute__((ext_vector_type(16))) __bf16 v16bf;
typedef __attribute__((ext_vector_type(8)))  __bf16 v8bf;
typedef __attribute__((ext_vector_type(8)))  float  v8f;

typedef unsigned short u16;
typedef unsigned int   u32;

// LDS row padding (in u16 elements) to spread rows across banks
#define XS_STRIDE (Dd + 8)   // 1032 u16 -> 2064 B per row (16B aligned)
#define HS_STRIDE (Hh + 8)   // 2056 u16 -> 4112 B per row (16B aligned)

__device__ __forceinline__ u16 f2bf(float f) {
    union { float f; u32 u; } v; v.f = f;
    u32 u = v.u;
    u += 0x7FFFu + ((u >> 16) & 1u);   // round-to-nearest-even
    return (u16)(u >> 16);
}

__device__ __forceinline__ v16bf cat8(v8bf lo, v8bf hi) {
    return __builtin_shufflevector(lo, hi, 0,1,2,3,4,5,6,7,8,9,10,11,12,13,14,15);
}

#define WMMA_BF16(A, B, C) \
    __builtin_amdgcn_wmma_f32_16x16x32_bf16(false, (A), false, (B), (short)0, (C), false, false)

#define V8F_ZERO (v8f){0.f,0.f,0.f,0.f,0.f,0.f,0.f,0.f}

// ---------------------------------------------------------------------------
// Prep: convert f32 weights to bf16 in WMMA-B-fragment-swizzled layout.
// Fragment f = (e*n16_per + n16)*ks_per + ks holds the 32x16 (KxN) tile:
//   lane L (N col L%16, h=L/16) stores 16 K values: k=8h+s (s<8), 16+8h+s-8 (s>=8).
// ---------------------------------------------------------------------------
__global__ __launch_bounds__(256)
void prep_swizzle(const float* __restrict__ src, u16* __restrict__ dst,
                  int n16_per, int ks_per) {
    const int g    = blockIdx.x * 256 + threadIdx.x;
    const int frag = g >> 5;
    const int lane = g & 31;
    const int per_e = n16_per * ks_per;
    const int e   = frag / per_e;
    const int r   = frag % per_e;
    const int n16 = r / ks_per;
    const int ks  = r % ks_per;
    const int N = n16_per * 16;
    const int K = ks_per * 32;

    const float* s = src + (size_t)(e * K + ks * 32 + (lane >> 4) * 8) * N + n16 * 16 + (lane & 15);

    uint4 o0, o1;
    o0.x = (u32)f2bf(s[0 * N])  | ((u32)f2bf(s[1 * N])  << 16);
    o0.y = (u32)f2bf(s[2 * N])  | ((u32)f2bf(s[3 * N])  << 16);
    o0.z = (u32)f2bf(s[4 * N])  | ((u32)f2bf(s[5 * N])  << 16);
    o0.w = (u32)f2bf(s[6 * N])  | ((u32)f2bf(s[7 * N])  << 16);
    o1.x = (u32)f2bf(s[16 * N]) | ((u32)f2bf(s[17 * N]) << 16);
    o1.y = (u32)f2bf(s[18 * N]) | ((u32)f2bf(s[19 * N]) << 16);
    o1.z = (u32)f2bf(s[20 * N]) | ((u32)f2bf(s[21 * N]) << 16);
    o1.w = (u32)f2bf(s[22 * N]) | ((u32)f2bf(s[23 * N]) << 16);

    uint4* d = (uint4*)dst + (size_t)frag * 64 + lane * 2;   // 1024 B per frag
    d[0] = o0;
    d[1] = o1;
}

// ---------------------------------------------------------------------------
// Gate: G[t,e] = x[t,:] . Wg[:,e] + bg[e]   (tiny: 0.13 GFLOP)
// ---------------------------------------------------------------------------
__global__ __launch_bounds__(256)
void gate_kernel(const float* __restrict__ x, const float* __restrict__ Wg,
                 const float* __restrict__ bg, float* __restrict__ G) {
    const int g = blockIdx.x * 256 + threadIdx.x;     // 65536 = T*E
    const int t = g >> 3;
    const int e = g & 7;
    float acc = bg[e];
    const float* xp = x + (size_t)t * Dd;
    for (int d = 0; d < Dd; ++d)
        acc = fmaf(xp[d], Wg[d * Ee + e], acc);
    G[g] = acc;
}

// ---------------------------------------------------------------------------
// Fused main kernel: per block = 32 tokens x full D output.
//   per expert: h = relu(x@W1+b1) -> LDS bf16 ; acc += gate*(h@W2+b2)
// ---------------------------------------------------------------------------
__global__ __launch_bounds__(NTHR)
void moe_main(const float* __restrict__ x,  const float* __restrict__ b1,
              const float* __restrict__ b2, const float* __restrict__ gate,
              const u16* __restrict__ w1s,  const u16* __restrict__ w2s,
              float* __restrict__ out) {
    extern __shared__ char smem[];
    u16*   xs = (u16*)smem;                                   // 32 x XS_STRIDE
    u16*   hs = (u16*)(smem + TM * XS_STRIDE * sizeof(u16));  // 32 x HS_STRIDE
    float* gl = (float*)(smem + TM * XS_STRIDE * sizeof(u16)
                              + TM * HS_STRIDE * sizeof(u16)); // 32 x 8

    const int tid   = threadIdx.x;
    const int lane  = tid & 31;
    const int w     = tid >> 5;              // wave id, 0..15
    const int tok0  = blockIdx.x * TM;
    const int half  = (lane >= 16) ? 8 : 0;  // A/C fragment row-half select
    const int ncol  = lane & 15;             // C fragment column

    // ---- stage x tile (f32 -> bf16, packed pair stores) into LDS ----
    for (int i = tid; i < TM * Dd / 2; i += NTHR) {
        const int row  = i >> 9;             // / 512 pairs per row
        const int colp = i & 511;
        const float2 v = *(const float2*)(x + (size_t)(tok0 + row) * Dd + colp * 2);
        *(u32*)(xs + row * XS_STRIDE + colp * 2) =
            (u32)f2bf(v.x) | ((u32)f2bf(v.y) << 16);
    }
    // ---- stage gates ----
    if (tid < TM * Ee)
        gl[tid] = gate[(size_t)(tok0 + (tid >> 3)) * Ee + (tid & 7)];
    __syncthreads();

    // persistent gate-weighted output accumulators: 4 col-tiles x 2 row-tiles
    v8f acc[8];
    #pragma unroll
    for (int i = 0; i < 8; ++i) acc[i] = V8F_ZERO;

    for (int e = 0; e < Ee; ++e) {
        // ================= GEMM1: h = relu(x @ W1[e] + b1[e]) =============
        // wave owns h col-tiles [w*8, w*8+8), processed 4 at a time
        for (int cp = 0; cp < 2; ++cp) {
            const int n16 = w * 8 + cp * 4;       // tiles n16 .. n16+3
            v8f h[8];                             // [col j][rowtile]
            #pragma unroll
            for (int i = 0; i < 8; ++i) h[i] = V8F_ZERO;

            const v8bf* wp0 = (const v8bf*)w1s
                            + (size_t)((e * 128 + n16) * 32) * 64 + lane * 2;
            const v8bf* wp1 = wp0 + (size_t)1 * 32 * 64;
            const v8bf* wp2 = wp0 + (size_t)2 * 32 * 64;
            const v8bf* wp3 = wp0 + (size_t)3 * 32 * 64;
            const u16* a0p = xs + (lane & 15) * XS_STRIDE + half;
            const u16* a1p = a0p + 16 * XS_STRIDE;

            for (int ks = 0; ks < 32; ++ks) {
                v16bf a0 = cat8(*(const v8bf*)(a0p), *(const v8bf*)(a0p + 16));
                v16bf a1 = cat8(*(const v8bf*)(a1p), *(const v8bf*)(a1p + 16));
                a0p += 32; a1p += 32;
                v16bf b0 = cat8(wp0[0], wp0[1]); wp0 += 64;
                v16bf b1v = cat8(wp1[0], wp1[1]); wp1 += 64;
                v16bf b2v = cat8(wp2[0], wp2[1]); wp2 += 64;
                v16bf b3v = cat8(wp3[0], wp3[1]); wp3 += 64;
                h[0] = WMMA_BF16(a0, b0,  h[0]);  h[1] = WMMA_BF16(a1, b0,  h[1]);
                h[2] = WMMA_BF16(a0, b1v, h[2]);  h[3] = WMMA_BF16(a1, b1v, h[3]);
                h[4] = WMMA_BF16(a0, b2v, h[4]);  h[5] = WMMA_BF16(a1, b2v, h[5]);
                h[6] = WMMA_BF16(a0, b3v, h[6]);  h[7] = WMMA_BF16(a1, b3v, h[7]);
            }
            #pragma unroll
            for (int j = 0; j < 4; ++j) {
                const float bias = b1[e * Hh + (n16 + j) * 16 + ncol];
                #pragma unroll
                for (int r = 0; r < 8; ++r) {
                    const int row0 = (r + half) * HS_STRIDE;
                    const int row1 = (16 + r + half) * HS_STRIDE;
                    hs[row0 + (n16 + j) * 16 + ncol] =
                        f2bf(fmaxf(h[j * 2 + 0][r] + bias, 0.f));
                    hs[row1 + (n16 + j) * 16 + ncol] =
                        f2bf(fmaxf(h[j * 2 + 1][r] + bias, 0.f));
                }
            }
        }
        __syncthreads();

        // ========= GEMM2: acc += gate[:,e] * (h @ W2[e] + b2[e]) ==========
        // wave owns out col-tiles [w*4, w*4+4), processed 4 at a time
        {
            const int d16 = w * 4;                // tiles d16 .. d16+3
            v8f o[8];                             // [col j][rowtile]
            #pragma unroll
            for (int i = 0; i < 8; ++i) o[i] = V8F_ZERO;

            const v8bf* wp0 = (const v8bf*)w2s
                            + (size_t)((e * 64 + d16) * 64) * 64 + lane * 2;
            const v8bf* wp1 = wp0 + (size_t)1 * 64 * 64;
            const v8bf* wp2 = wp0 + (size_t)2 * 64 * 64;
            const v8bf* wp3 = wp0 + (size_t)3 * 64 * 64;
            const u16* a0p = hs + (lane & 15) * HS_STRIDE + half;
            const u16* a1p = a0p + 16 * HS_STRIDE;

            for (int ks = 0; ks < 64; ++ks) {
                v16bf a0 = cat8(*(const v8bf*)(a0p), *(const v8bf*)(a0p + 16));
                v16bf a1 = cat8(*(const v8bf*)(a1p), *(const v8bf*)(a1p + 16));
                a0p += 32; a1p += 32;
                v16bf b0 = cat8(wp0[0], wp0[1]); wp0 += 64;
                v16bf b1v = cat8(wp1[0], wp1[1]); wp1 += 64;
                v16bf b2v = cat8(wp2[0], wp2[1]); wp2 += 64;
                v16bf b3v = cat8(wp3[0], wp3[1]); wp3 += 64;
                o[0] = WMMA_BF16(a0, b0,  o[0]);  o[1] = WMMA_BF16(a1, b0,  o[1]);
                o[2] = WMMA_BF16(a0, b1v, o[2]);  o[3] = WMMA_BF16(a1, b1v, o[3]);
                o[4] = WMMA_BF16(a0, b2v, o[4]);  o[5] = WMMA_BF16(a1, b2v, o[5]);
                o[6] = WMMA_BF16(a0, b3v, o[6]);  o[7] = WMMA_BF16(a1, b3v, o[7]);
            }
            #pragma unroll
            for (int j = 0; j < 4; ++j) {
                const float bias2 = b2[e * Dd + (d16 + j) * 16 + ncol];
                #pragma unroll
                for (int r = 0; r < 8; ++r) {
                    const float g0 = gl[(r + half) * Ee + e];
                    const float g1 = gl[(16 + r + half) * Ee + e];
                    acc[j * 2 + 0][r] += g0 * (o[j * 2 + 0][r] + bias2);
                    acc[j * 2 + 1][r] += g1 * (o[j * 2 + 1][r] + bias2);
                }
            }
        }
        __syncthreads();   // hs reused by next expert
    }

    // ---- write 32 x 1024 f32 output tile ----
    for (int c = 0; c < 4; ++c) {
        const int d16 = w * 4 + c;
        #pragma unroll
        for (int r = 0; r < 8; ++r) {
            out[(size_t)(tok0 + r + half) * Dd + d16 * 16 + ncol]      = acc[c * 2 + 0][r];
            out[(size_t)(tok0 + 16 + r + half) * Dd + d16 * 16 + ncol] = acc[c * 2 + 1][r];
        }
    }
}

// ---------------------------------------------------------------------------
extern "C" void kernel_launch(void* const* d_in, const int* in_sizes, int n_in,
                              void* d_out, int out_size, void* d_ws, size_t ws_size,
                              hipStream_t stream) {
    const float* x  = (const float*)d_in[0];
    const float* W1 = (const float*)d_in[1];
    const float* b1 = (const float*)d_in[2];
    const float* W2 = (const float*)d_in[3];
    const float* b2 = (const float*)d_in[4];
    const float* Wg = (const float*)d_in[5];
    const float* bg = (const float*)d_in[6];
    float* out = (float*)d_out;

    // workspace layout: bf16-swizzled W1 (32 MB), W2 (32 MB), gates (256 KB)
    u16*   w1s = (u16*)d_ws;
    u16*   w2s = (u16*)((char*)d_ws + (size_t)Ee * Dd * Hh * 2);
    float* G   = (float*)((char*)d_ws + (size_t)2 * Ee * Dd * Hh * 2);

    // W1: [E][D=1024][H=2048] -> n16_per = 128, ks_per = 32
    prep_swizzle<<<4096, 256, 0, stream>>>(W1, w1s, 128, 32);
    // W2: [E][H=2048][D=1024] -> n16_per = 64, ks_per = 64
    prep_swizzle<<<4096, 256, 0, stream>>>(W2, w2s, 64, 64);
    // gates
    gate_kernel<<<(Tt * Ee) / 256, 256, 0, stream>>>(x, Wg, bg, G);

    const size_t smem = (size_t)TM * XS_STRIDE * 2 + (size_t)TM * HS_STRIDE * 2
                      + (size_t)TM * Ee * 4;   // ~198 KB dynamic LDS (<320 KB/WGP)
    moe_main<<<Tt / TM, NTHR, smem, stream>>>(x, b1, b2, G, w1s, w2s, out);
}
